// CaptionHead_15195594293897
// MI455X (gfx1250) — matrix operand order repaired and checked by reading, not compiled, not run on hardware
//
#include <hip/hip_runtime.h>
#include <hip/hip_bf16.h>

typedef __bf16 bf16_t;
typedef __attribute__((ext_vector_type(16))) __bf16 v16bf;
typedef __attribute__((ext_vector_type(8)))  __bf16 v8bf;
typedef __attribute__((ext_vector_type(8)))  float  v8f;
typedef __attribute__((ext_vector_type(2)))  float  v2f;
typedef __attribute__((ext_vector_type(4)))  unsigned int v4u;
typedef __attribute__((ext_vector_type(4)))  int    v4i;
typedef __attribute__((ext_vector_type(8)))  int    v8i;

#define C_DIM   512
#define P_PTS   131072
#define M_CAP   1024
#define PORIG   163840
#define T_PAIR  262144

// ---------------------------------------------------------------- ptom setup
__global__ void k_ptom_init(int* __restrict__ ptom) {
    int i = blockIdx.x * blockDim.x + threadIdx.x;
    if (i < PORIG) ptom[i] = -1;
}

__global__ void k_ptom_scatter(int* __restrict__ ptom, const int* __restrict__ oidx) {
    int i = blockIdx.x * blockDim.x + threadIdx.x;
    if (i < P_PTS) ptom[oidx[i]] = i;
}

// ------------------------------------------------- caption embed f32 -> bf16
__global__ void k_capcvt(const float* __restrict__ cap, bf16_t* __restrict__ capb) {
    int i = blockIdx.x * blockDim.x + threadIdx.x;
    capb[i] = (bf16_t)cap[i];
}

// ---------------------------- gather voxel row, L2-normalize, store bf16 row
__global__ __launch_bounds__(256)
void k_normalize(const float* __restrict__ adapter, const int* __restrict__ v2p,
                 bf16_t* __restrict__ feats) {
    int wave = threadIdx.x >> 5;
    int lane = threadIdx.x & 31;
    int p = blockIdx.x * 8 + wave;                 // one wave per point
    const float* src = adapter + (size_t)v2p[p] * C_DIM + lane * 16;
    float4 x0 = ((const float4*)src)[0];
    float4 x1 = ((const float4*)src)[1];
    float4 x2 = ((const float4*)src)[2];
    float4 x3 = ((const float4*)src)[3];
    float v[16] = {x0.x,x0.y,x0.z,x0.w, x1.x,x1.y,x1.z,x1.w,
                   x2.x,x2.y,x2.z,x2.w, x3.x,x3.y,x3.z,x3.w};
    float ss = 0.f;
#pragma unroll
    for (int i = 0; i < 16; ++i) ss += v[i] * v[i];
#pragma unroll
    for (int m = 16; m; m >>= 1) ss += __shfl_xor(ss, m, 32);
    float inv = 1.0f / fmaxf(sqrtf(ss), 1e-12f);
    unsigned int w[8];
#pragma unroll
    for (int i = 0; i < 8; ++i) {
        unsigned short lo = __builtin_bit_cast(unsigned short, (bf16_t)(v[2*i]   * inv));
        unsigned short hi = __builtin_bit_cast(unsigned short, (bf16_t)(v[2*i+1] * inv));
        w[i] = (unsigned int)lo | ((unsigned int)hi << 16);
    }
    uint4* dst = (uint4*)(feats + (size_t)p * C_DIM + lane * 16);
    dst[0] = make_uint4(w[0], w[1], w[2], w[3]);
    dst[1] = make_uint4(w[4], w[5], w[6], w[7]);
}

// -------------------------------------------- Tensor Data Mover: 2D bf16 tile
__device__ __forceinline__ void tdm_load_2d(const void* gsrc, unsigned int lds_off,
                                            unsigned int tile0, unsigned int tile1,
                                            unsigned int tdim0, unsigned int tdim1,
                                            unsigned int stride0) {
    unsigned long long ga = (unsigned long long)(uintptr_t)gsrc;
    v4u g0;
    g0[0] = 1u;                                            // count=1, user mode
    g0[1] = lds_off;                                       // lds_addr
    g0[2] = (unsigned int)ga;                              // global_addr[31:0]
    g0[3] = (unsigned int)((ga >> 32) & 0x1FFFFFFu) | (2u << 30); // ga[56:32] | type=2
    v8i g1;
    g1[0] = (int)(1u << 16);                               // data_size=2 bytes
    g1[1] = (int)((tdim0 & 0xFFFFu) << 16);                // tensor_dim0 lo16
    g1[2] = (int)((tdim0 >> 16) | ((tdim1 & 0xFFFFu) << 16)); // tdim0 hi | tdim1 lo
    g1[3] = (int)((tdim1 >> 16) | (tile0 << 16));          // tdim1 hi | tile_dim0
    g1[4] = (int)(tile1 & 0xFFFFu);                        // tile_dim1 (tile_dim2=0)
    g1[5] = (int)stride0;                                  // tensor_dim0_stride lo32
    g1[6] = 0;                                             // stride0 hi | stride1 lo
    g1[7] = 0;
    v4i z4 = {0, 0, 0, 0};
    v8i z8 = {0, 0, 0, 0, 0, 0, 0, 0};
    __builtin_amdgcn_tensor_load_to_lds(g0, g1, z4, z4, z8, 0);
}

__device__ __forceinline__ unsigned int lds_off_of(const void* p) {
    return (unsigned int)(uintptr_t)p;   // generic LDS addr: low 32 bits = offset
}

// ------------------------------------------- big matmul reduced to lse[p]
// block: 8 waves, 128 points. A strips pinned in VGPRs; A tile + double-
// buffered B chunks delivered by the Tensor Data Mover (TENSORcnt pipelined).
// K loop split into two independent accumulator chains so B-fragment LDS
// loads for pair kp+1 overlap the WMMAs of pair kp.
__global__ __launch_bounds__(256)
void k_lse(const bf16_t* __restrict__ feats, const bf16_t* __restrict__ capb,
           const float* __restrict__ ls, float* __restrict__ lse) {
    __shared__ bf16_t ldsA[128 * C_DIM];       // 128 KB
    __shared__ bf16_t ldsB[2][16 * C_DIM];     // 2 x 16 KB
    const int tid  = threadIdx.x;
    const int wave = tid >> 5;
    const int lane = tid & 31;
    const int p0   = blockIdx.x * 128;

    // wave 0: TDM A tile (128x512 bf16 as 4096x16 2D) then B chunk 0.
    // TENSORcnt is in-order per wave: wait<=1 guarantees A has landed while
    // B0 may still be in flight.
    if (wave == 0) {
        tdm_load_2d(feats + (size_t)p0 * C_DIM, lds_off_of(ldsA),
                    4096u, 16u, 4096u, 16u, 4096u);
        tdm_load_2d(capb, lds_off_of(&ldsB[0][0]),
                    8192u, 1u, 8192u, 1u, 8192u);
        __builtin_amdgcn_s_wait_tensorcnt(1);
    }
    __syncthreads();

    // pin this wave's 16-point A strip in registers (16 k-chunks = 128 VGPRs)
    const int arow  = wave * 16 + (lane & 15);
    const int ahalf = (lane >> 4) * 8;          // A layout: K 0-7/16-23 vs 8-15/24-31
    v16bf aT[16];
#pragma unroll
    for (int kc = 0; kc < 16; ++kc) {
        const bf16_t* pa = &ldsA[arow * C_DIM + kc * 32 + ahalf];
        v8bf alo = *(const v8bf*)pa;
        v8bf ahi = *(const v8bf*)(pa + 16);
#pragma unroll
        for (int i = 0; i < 8; ++i) { aT[kc][i] = alo[i]; aT[kc][i + 8] = ahi[i]; }
    }

    const float scl = __expf(ls[0]);
    float se[8];
#pragma unroll
    for (int i = 0; i < 8; ++i) se[i] = 0.f;

    const int boff = (lane & 15) * C_DIM + (lane >> 4) * 8; // B: col=lane&15, K 0-7/16-23 vs 8-15/24-31

    for (int nc = 0; nc < 64; ++nc) {           // 16 captions per chunk
        if (wave == 0) __builtin_amdgcn_s_wait_tensorcnt(0);  // chunk nc in buf[nc&1]
        __syncthreads();
        if (wave == 0 && nc + 1 < 64)           // prefetch nc+1 into the other buffer
            tdm_load_2d(capb + (size_t)(nc + 1) * 16 * C_DIM,
                        lds_off_of(&ldsB[(nc + 1) & 1][0]),
                        8192u, 1u, 8192u, 1u, 8192u);

        const bf16_t* bbase = &ldsB[nc & 1][0];
        // two independent accumulation chains (even/odd k-chunks) with a
        // rotating two-pair B register file: loads run ahead of the WMMAs.
        v16bf b0, b1;
#pragma unroll
        for (int i = 0; i < 2; ++i) {
            v8bf blo = *(const v8bf*)&bbase[boff + i * 32];
            v8bf bhi = *(const v8bf*)&bbase[boff + i * 32 + 16];
            v16bf t;
#pragma unroll
            for (int j = 0; j < 8; ++j) { t[j] = blo[j]; t[j + 8] = bhi[j]; }
            if (i == 0) b0 = t; else b1 = t;
        }
        v8f c0 = {}, c1 = {};
#pragma unroll
        for (int kp = 0; kp < 8; ++kp) {
            v16bf n0, n1;
            if (kp < 7) {
#pragma unroll
                for (int i = 0; i < 2; ++i) {
                    v8bf blo = *(const v8bf*)&bbase[boff + (2*kp + 2 + i) * 32];
                    v8bf bhi = *(const v8bf*)&bbase[boff + (2*kp + 2 + i) * 32 + 16];
                    v16bf t;
#pragma unroll
                    for (int j = 0; j < 8; ++j) { t[j] = blo[j]; t[j + 8] = bhi[j]; }
                    if (i == 0) n0 = t; else n1 = t;
                }
            }
            c0 = __builtin_amdgcn_wmma_f32_16x16x32_bf16(
                     false, aT[2*kp],     false, b0, (short)0, c0, false, false);
            c1 = __builtin_amdgcn_wmma_f32_16x16x32_bf16(
                     false, aT[2*kp + 1], false, b1, (short)0, c1, false, false);
            b0 = n0; b1 = n1;
        }
        // logits bounded by |scale| (unit rows) -> safe unshifted sum of exp
#pragma unroll
        for (int i = 0; i < 8; ++i) se[i] += __expf(scl * (c0[i] + c1[i]));
    }

    // reduce the 16 column-lanes of each row group
#pragma unroll
    for (int i = 0; i < 8; ++i) {
#pragma unroll
        for (int m = 8; m; m >>= 1) se[i] += __shfl_xor(se[i], m, 32);
    }
    if (lane == 0) {
#pragma unroll
        for (int i = 0; i < 8; ++i) lse[p0 + wave * 16 + i] = __logf(se[i]);
    }
    if (lane == 16) {
#pragma unroll
        for (int i = 0; i < 8; ++i) lse[p0 + wave * 16 + 8 + i] = __logf(se[i]);
    }
}

// ------------------------------------------------ sorted-segment reduction
__device__ __forceinline__ int lower_bound(const int* a, int n, int key) {
    int lo = 0, hi = n;
    while (lo < hi) { int mid = (lo + hi) >> 1; if (a[mid] < key) lo = mid + 1; else hi = mid; }
    return lo;
}

__global__ __launch_bounds__(256)
void k_segreduce(const bf16_t* __restrict__ feats, const float* __restrict__ lse,
                 const int* __restrict__ ptom, const int* __restrict__ ctpm,
                 const int* __restrict__ seg, float* __restrict__ S,
                 float* __restrict__ L, float* __restrict__ rn,
                 float* __restrict__ out_tail) {
    const int c = blockIdx.x;
    const int tid = threadIdx.x;
    const int lo = lower_bound(seg, T_PAIR, c);
    const int hi = lower_bound(seg, T_PAIR, c + 1);
    float s0 = 0.f, s1 = 0.f, Lc = 0.f;
    int ninv = 0;
    int g = (lo < hi) ? ptom[ctpm[lo]] : 0;       // broadcast load
    for (int t = lo; t < hi; ++t) {
        int gn = (t + 1 < hi) ? ptom[ctpm[t + 1]] : 0;
        int p = g < 0 ? g + P_PTS : g;            // torch negative-index wrap
        int pn = gn < 0 ? gn + P_PTS : gn;
        __builtin_prefetch(feats + (size_t)pn * C_DIM + 2 * tid, 0, 0);
        unsigned int pair = *(const unsigned int*)(feats + (size_t)p * C_DIM + 2 * tid);
        s0 += (float)__builtin_bit_cast(bf16_t, (unsigned short)(pair & 0xFFFFu));
        s1 += (float)__builtin_bit_cast(bf16_t, (unsigned short)(pair >> 16));
        if (tid == 0) { Lc += lse[p]; ninv += (g < 0); }
        g = gn;
    }
    S[(size_t)c * C_DIM + 2 * tid]     = s0;
    S[(size_t)c * C_DIM + 2 * tid + 1] = s1;
    if (tid == 0) {
        L[c] = Lc;
        float n = (float)(hi - lo - ninv);
        rn[c] = n;
        out_tail[c]         = n;                  // real_n_points
        out_tail[M_CAP + c] = n > 0.f ? 1.0f : 0.0f;  // if_has_pts
    }
}

// ------------------------- pooled = (scale * S @ E^T - L) / rn  (f32 WMMA)
__global__ __launch_bounds__(256)
void k_final(const float* __restrict__ S, const float* __restrict__ E,
             const float* __restrict__ L, const float* __restrict__ rn,
             const float* __restrict__ ls, float* __restrict__ out) {
    const int wave = threadIdx.x >> 5;
    const int lane = threadIdx.x & 31;
    const int tile = blockIdx.x * 8 + wave;       // 64x64 tiles of 16x16
    const int tc = tile >> 6, tm = tile & 63;
    const int r = lane & 15, g = lane >> 4;
    const float* sa = S + (size_t)(tc * 16 + r) * C_DIM + g * 2;
    const float* sb = E + (size_t)(tm * 16 + r) * C_DIM + g * 2;
    v8f acc = {};
#pragma unroll 8
    for (int kk = 0; kk < 128; ++kk) {
        v2f a = *(const v2f*)(sa + kk * 4);
        v2f b = *(const v2f*)(sb + kk * 4);
        acc = __builtin_amdgcn_wmma_f32_16x16x4_f32(
                  false, a, false, b, (short)0, acc, false, false);
    }
    const float scl = __expf(ls[0]);
    const int m = tm * 16 + r;
#pragma unroll
    for (int i = 0; i < 8; ++i) {
        int c = tc * 16 + i + g * 8;
        float n = rn[c];
        float denom = n > 0.f ? 1.0f / n : 0.f;
        out[(size_t)c * M_CAP + m] = (scl * acc[i] - L[c]) * denom;
    }
}

// ---------------------------------------------------------------- launcher
extern "C" void kernel_launch(void* const* d_in, const int* in_sizes, int n_in,
                              void* d_out, int out_size, void* d_ws, size_t ws_size,
                              hipStream_t stream) {
    (void)in_sizes; (void)n_in; (void)out_size; (void)ws_size;
    const float* adapter = (const float*)d_in[0];
    const float* cap     = (const float*)d_in[1];
    const float* ls      = (const float*)d_in[2];
    const int*   v2p     = (const int*)d_in[3];
    const int*   oidx    = (const int*)d_in[4];
    const int*   ctpm    = (const int*)d_in[5];
    const int*   seg     = (const int*)d_in[6];

    char* ws = (char*)d_ws;
    bf16_t* feats = (bf16_t*)(ws);                         // 134,217,728 B
    bf16_t* capb  = (bf16_t*)(ws + 134217728);             //   1,048,576 B
    float*  lse   = (float*) (ws + 135266304);             //     524,288 B
    int*    ptom  = (int*)   (ws + 135790592);             //     655,360 B
    float*  S     = (float*) (ws + 136445952);             //   2,097,152 B
    float*  L     = (float*) (ws + 138543104);             //       4,096 B
    float*  rn    = (float*) (ws + 138547200);             //       4,096 B
    float*  out   = (float*)d_out;

    k_ptom_init   <<<PORIG / 256, 256, 0, stream>>>(ptom);
    k_ptom_scatter<<<P_PTS / 256, 256, 0, stream>>>(ptom, oidx);
    k_capcvt      <<<(M_CAP * C_DIM) / 256, 256, 0, stream>>>(cap, capb);
    k_normalize   <<<P_PTS / 8, 256, 0, stream>>>(adapter, v2p, feats);
    k_lse         <<<P_PTS / 128, 256, 0, stream>>>(feats, capb, ls, lse);
    k_segreduce   <<<M_CAP, 256, 0, stream>>>(feats, lse, ptom, ctpm, seg,
                                              S, L, rn, out + M_CAP * M_CAP);
    k_final       <<<(64 * 64) / 8, 256, 0, stream>>>(S, cap, L, rn, ls, out);
}
